// NGCFEmbedding_73976516706690
// MI455X (gfx1250) — matrix operand order repaired and checked by reading, not compiled, not run on hardware
//
#include <hip/hip_runtime.h>

// ---------------------------------------------------------------------------
// NGCF bipartite GNN for MI455X (gfx1250, wave32, WMMA).
// Matmuls on v_wmma_f32_16x16x32_bf16 (fp32 accumulate); W staged into LDS
// via global_load_async_to_lds_b128 (ASYNCcnt); B fragments double-buffered
// so ds_load latency overlaps the matrix pipe; scatters via
// global_atomic_add_f32.
// ---------------------------------------------------------------------------

typedef __attribute__((ext_vector_type(16))) __bf16 v16bf;
typedef __attribute__((ext_vector_type(8)))  float  v8f;

#define DEV static __device__ __forceinline__

// round-to-nearest-even f32 -> bf16 (as u16)
DEV unsigned short f2bf(float f) {
  unsigned u = __float_as_uint(f);
  return (unsigned short)((u + 0x7fffu + ((u >> 16) & 1u)) >> 16);
}

// pack two f32 into one u32 of two bf16 (lo, hi), RNE
DEV unsigned int pack_bf16_rne(float lo, float hi) {
  unsigned ul = __float_as_uint(lo);
  ul = (ul + 0x7fffu + ((ul >> 16) & 1u)) >> 16;
  unsigned uh = __float_as_uint(hi);
  uh = (uh + 0x7fffu + ((uh >> 16) & 1u)) & 0xffff0000u;
  return (ul & 0xffffu) | uh;
}

// elementwise product of two bf16 pairs (packed u32), result packed bf16
DEV unsigned int bfmul2(unsigned int a, unsigned int b) {
  float al = __uint_as_float(a << 16);
  float ah = __uint_as_float(a & 0xffff0000u);
  float bl = __uint_as_float(b << 16);
  float bh = __uint_as_float(b & 0xffff0000u);
  return pack_bf16_rne(al * bl, ah * bh);
}

// hardware f32 global atomic add, no return (STOREcnt tracked; s_endpgm drains)
DEV void atomic_add_f32(float* p, float v) {
  asm volatile("global_atomic_add_f32 %0, %1, off" :: "v"(p), "v"(v) : "memory");
}

// async 16B copy global -> LDS (ASYNCcnt tracked, bypasses VGPRs).
// lds_off: LDS byte address (low 32 bits of generic shared pointer).
DEV void async_g2lds_16B(unsigned lds_off, const void* gptr) {
  asm volatile("global_load_async_to_lds_b128 %0, %1, off"
               :: "v"(lds_off), "v"(gptr) : "memory");
}
DEV void wait_asynccnt0() {
  asm volatile("s_wait_asynccnt 0x0" ::: "memory");
}

// block-wide: stage 128x128 bf16 (32KB) weight matrix into LDS
DEV void stage_W(uint4* sW, const unsigned short* Wbf) {
  unsigned ldsBase = (unsigned)(size_t)(void*)sW;  // low 32 bits = LDS offset
  const char* g = (const char*)Wbf;
  for (int i = threadIdx.x; i < 2048; i += 256)
    async_g2lds_16B(ldsBase + (unsigned)i * 16u, g + (size_t)i * 16);
  wait_asynccnt0();
  __syncthreads();
}

// ---------------------------------------------------------------------------
// small elementwise kernels
// ---------------------------------------------------------------------------

__global__ void k_degree(const int* __restrict__ s, const int* __restrict__ d,
                         unsigned int* __restrict__ degU,
                         unsigned int* __restrict__ degI, int E) {
  int i = blockIdx.x * blockDim.x + threadIdx.x;
  if (i < E) {
    atomicAdd(&degU[s[i]], 1u);
    atomicAdd(&degI[d[i]], 1u);
  }
}

__global__ void k_weight(const int* __restrict__ s, const int* __restrict__ d,
                         const unsigned int* __restrict__ degU,
                         const unsigned int* __restrict__ degI,
                         float* __restrict__ wE, int E) {
  int i = blockIdx.x * blockDim.x + threadIdx.x;
  if (i < E) {
    float p = (float)degU[s[i]] * (float)degI[d[i]];
    wE[i] = rsqrtf(fmaxf(p, 1.0f));
  }
}

// X0 = emb ; out = emb/3   (both sides)
__global__ void k_init(const float* __restrict__ ue, const float* __restrict__ ie,
                       float* __restrict__ Xu, float* __restrict__ Xi,
                       float* __restrict__ oU, float* __restrict__ oI, int n) {
  int i = blockIdx.x * blockDim.x + threadIdx.x;
  if (i < n) {
    float u = ue[i], t = ie[i];
    Xu[i] = u; Xi[i] = t;
    oU[i] = u * (1.0f / 3.0f);
    oI[i] = t * (1.0f / 3.0f);
  }
}

// bf16 conversion of two arrays
__global__ void k_cvt2(const float* __restrict__ a, const float* __restrict__ b,
                       unsigned short* __restrict__ A, unsigned short* __restrict__ B,
                       int n) {
  int i = blockIdx.x * blockDim.x + threadIdx.x;
  if (i < n) { A[i] = f2bf(a[i]); B[i] = f2bf(b[i]); }
}

// Xnext = leaky_relu(T + Agg) ; out += Xnext/3
__global__ void k_combine(const float* __restrict__ T, const float* __restrict__ Agg,
                          float* __restrict__ Xn, float* __restrict__ outAcc, int n) {
  int i = blockIdx.x * blockDim.x + threadIdx.x;
  if (i < n) {
    float v = T[i] + Agg[i];
    v = (v > 0.0f) ? v : 0.01f * v;
    Xn[i] = v;
    outAcc[i] += v * (1.0f / 3.0f);
  }
}

// ---------------------------------------------------------------------------
// B fragment: 32x16 bf16 tile of W^T; B[k][n] = W[n][k] so lane (lo,hi) reads
// 8 bf16 of W row (nt*16+lo) at k-offsets kt*32+hi*8 and kt*32+16+hi*8.
// ---------------------------------------------------------------------------
union BFrag { uint4 q[2]; v16bf v; };

DEV void load_B(BFrag& B, const unsigned short* sWl, int nt, int kt, int lo, int hi) {
  const unsigned short* bp =
      sWl + (size_t)(nt * 16 + lo) * 128 + kt * 32 + hi * 8;
  B.q[0] = *(const uint4*)bp;
  B.q[1] = *(const uint4*)(bp + 16);
}

// ---------------------------------------------------------------------------
// dense GEMM: T[r][:] = X[r][:] @ W^T   (X bf16 row-major, W bf16 row-major)
// one wave per 16-row tile; W (128x128 bf16 = 32KB) staged in LDS.
// ---------------------------------------------------------------------------
__global__ void __launch_bounds__(256) k_dense_wmma(
    const unsigned short* __restrict__ Xbf,
    const unsigned short* __restrict__ Wbf,
    float* __restrict__ T, int nRows) {
  __shared__ uint4 sW[2048];  // 128*128 bf16
  stage_W(sW, Wbf);

  int wave = (blockIdx.x * 256 + threadIdx.x) >> 5;
  int lane = threadIdx.x & 31;
  int row0 = wave << 4;
  if (row0 >= nRows) return;
  int lo = lane & 15, hi = lane >> 4;

  const unsigned short* xrow = Xbf + (size_t)(row0 + lo) * 128;
  const unsigned short* sWl  = (const unsigned short*)sW;

  v8f acc[8];
#pragma unroll
  for (int nt = 0; nt < 8; ++nt)
#pragma unroll
    for (int j = 0; j < 8; ++j) acc[nt][j] = 0.0f;

#pragma unroll
  for (int kt = 0; kt < 4; ++kt) {
    // A fragment: 16x32 bf16, ISA layout (lanes0-15: K 0-7 & 16-23 of kt*32)
    union { uint4 q[2]; v16bf v; } A;
    A.q[0] = *(const uint4*)(xrow + kt * 32 + hi * 8);
    A.q[1] = *(const uint4*)(xrow + kt * 32 + 16 + hi * 8);

    // double-buffered B: ds_load for nt+1 issues before wmma for nt
    BFrag B[2];
    load_B(B[0], sWl, 0, kt, lo, hi);
#pragma unroll
    for (int nt = 0; nt < 8; ++nt) {
      if (nt < 7) load_B(B[(nt + 1) & 1], sWl, nt + 1, kt, lo, hi);
      acc[nt] = __builtin_amdgcn_wmma_f32_16x16x32_bf16(
          false, A.v, false, B[nt & 1].v, (short)0, acc[nt], false, false);
    }
  }

  // C/D layout: VGPR g -> row g + 8*hi, col = nt*16 + lo
#pragma unroll
  for (int nt = 0; nt < 8; ++nt) {
    int c = nt * 16 + lo;
#pragma unroll
    for (int g = 0; g < 8; ++g)
      T[(size_t)(row0 + g + 8 * hi) * 128 + c] = acc[nt][g];
  }
}

// ---------------------------------------------------------------------------
// fused edge cell: for 16-edge tile, Q = (x_src ⊙ x_dst) @ W_inter^T (WMMA),
// then atomically scatter w[e]*(Tsrc[src]+Q) into agg[dst].
// ---------------------------------------------------------------------------
__global__ void __launch_bounds__(256) k_edge_wmma(
    const unsigned short* __restrict__ srcBf,
    const unsigned short* __restrict__ dstBf,
    const float* __restrict__ Tsrc,
    const int* __restrict__ srcIdx,
    const int* __restrict__ dstIdx,
    const float* __restrict__ wE,
    const unsigned short* __restrict__ Wbf,
    float* __restrict__ agg, int nEdges) {
  __shared__ uint4 sW[2048];  // W_inter bf16, 32KB
  stage_W(sW, Wbf);

  int wave = (blockIdx.x * 256 + threadIdx.x) >> 5;
  int lane = threadIdx.x & 31;
  int e0 = wave << 4;
  if (e0 >= nEdges) return;
  int lo = lane & 15, hi = lane >> 4;

  // this lane builds A-row (lo): the gathered product row of edge e0+lo
  int eA = e0 + lo;
  int sA = srcIdx[eA];
  const unsigned short* srow = srcBf + (size_t)sA * 128;
  const unsigned short* drow = dstBf + (size_t)dstIdx[eA] * 128;
  __builtin_prefetch(Tsrc + (size_t)sA * 128, 0, 1);  // global_prefetch_b8

  const unsigned short* sWl = (const unsigned short*)sW;

  v8f acc[8];
#pragma unroll
  for (int nt = 0; nt < 8; ++nt)
#pragma unroll
    for (int j = 0; j < 8; ++j) acc[nt][j] = 0.0f;

#pragma unroll
  for (int kt = 0; kt < 4; ++kt) {
    uint4 s1 = *(const uint4*)(srow + kt * 32 + hi * 8);
    uint4 d1 = *(const uint4*)(drow + kt * 32 + hi * 8);
    uint4 s2 = *(const uint4*)(srow + kt * 32 + 16 + hi * 8);
    uint4 d2 = *(const uint4*)(drow + kt * 32 + 16 + hi * 8);
    union { unsigned int u[8]; v16bf v; } A;
    A.u[0] = bfmul2(s1.x, d1.x);
    A.u[1] = bfmul2(s1.y, d1.y);
    A.u[2] = bfmul2(s1.z, d1.z);
    A.u[3] = bfmul2(s1.w, d1.w);
    A.u[4] = bfmul2(s2.x, d2.x);
    A.u[5] = bfmul2(s2.y, d2.y);
    A.u[6] = bfmul2(s2.z, d2.z);
    A.u[7] = bfmul2(s2.w, d2.w);

    BFrag B[2];
    load_B(B[0], sWl, 0, kt, lo, hi);
#pragma unroll
    for (int nt = 0; nt < 8; ++nt) {
      if (nt < 7) load_B(B[(nt + 1) & 1], sWl, nt + 1, kt, lo, hi);
      acc[nt] = __builtin_amdgcn_wmma_f32_16x16x32_bf16(
          false, A.v, false, B[nt & 1].v, (short)0, acc[nt], false, false);
    }
  }

  // epilogue: C row g+8*hi <-> edge e0+g+8*hi ; col = nt*16+lo
  int sg[8], dg[8]; float wg[8];
#pragma unroll
  for (int g = 0; g < 8; ++g) {
    int e = e0 + g + 8 * hi;
    sg[g] = srcIdx[e];
    dg[g] = dstIdx[e];
    wg[g] = wE[e];
  }
#pragma unroll
  for (int nt = 0; nt < 8; ++nt) {
    int c = nt * 16 + lo;
#pragma unroll
    for (int g = 0; g < 8; ++g) {
      float val = wg[g] * (Tsrc[(size_t)sg[g] * 128 + c] + acc[nt][g]);
      atomic_add_f32(&agg[(size_t)dg[g] * 128 + c], val);
    }
  }
}

// ---------------------------------------------------------------------------
// host orchestration
// ---------------------------------------------------------------------------
extern "C" void kernel_launch(void* const* d_in, const int* in_sizes, int n_in,
                              void* d_out, int out_size, void* d_ws, size_t ws_size,
                              hipStream_t stream) {
  (void)in_sizes; (void)n_in; (void)out_size; (void)ws_size;
  const int N = 150000, D = 128, E = 1000000;
  const int ND = N * D;

  // inputs: 0 usr_n_id, 1 itm_n_id (identity - ignored), 2 usr_edge_index,
  // 3 itm_edge_index (reverse of 2 - ignored), 4 usr_emb, 5 itm_emb, 6..9 W
  const int* usrE = (const int*)d_in[2];
  const int* eSrc = usrE;       // users   (usr_edge_index[0])
  const int* eDst = usrE + E;   // items   (usr_edge_index[1])
  const float* usr_emb = (const float*)d_in[4];
  const float* itm_emb = (const float*)d_in[5];
  const float* W[4] = {(const float*)d_in[6], (const float*)d_in[7],
                       (const float*)d_in[8], (const float*)d_in[9]};

  // workspace carve-up (256B aligned slices)
  char* ws = (char*)d_ws;
  size_t off = 0;
  auto take = [&](size_t bytes) {
    char* p = ws + off;
    off = (off + bytes + 255) & ~(size_t)255;
    return p;
  };
  unsigned int* degU = (unsigned int*)take((size_t)N * 4);
  unsigned int* degI = (unsigned int*)take((size_t)N * 4);
  float* wE   = (float*)take((size_t)E * 4);
  float* XuA  = (float*)take((size_t)ND * 4);
  float* XiA  = (float*)take((size_t)ND * 4);
  float* XuB  = (float*)take((size_t)ND * 4);
  float* XiB  = (float*)take((size_t)ND * 4);
  float* Tu   = (float*)take((size_t)ND * 4);
  float* Ti   = (float*)take((size_t)ND * 4);
  float* AggU = (float*)take((size_t)ND * 4);
  float* AggI = (float*)take((size_t)ND * 4);
  unsigned short* XuBf = (unsigned short*)take((size_t)ND * 2);
  unsigned short* XiBf = (unsigned short*)take((size_t)ND * 2);
  unsigned short* WaBf = (unsigned short*)take((size_t)D * D * 2);
  unsigned short* WbBf = (unsigned short*)take((size_t)D * D * 2);

  float* outU = (float*)d_out;
  float* outI = (float*)d_out + (size_t)ND;

  const int TPB = 256;
  const int gE  = (E + TPB - 1) / TPB;
  const int gND = (ND + TPB - 1) / TPB;
  const int gDD = (D * D + TPB - 1) / TPB;
  const int gDense = ((N / 16) + 7) / 8;   // 9375 tiles, 8 waves/block
  const int gEdge  = ((E / 16) + 7) / 8;   // 62500 tiles

  // edge weights (symmetric: itm direction has identical per-edge weight)
  hipMemsetAsync(degU, 0, (size_t)N * 4, stream);
  hipMemsetAsync(degI, 0, (size_t)N * 4, stream);
  k_degree<<<gE, TPB, 0, stream>>>(eSrc, eDst, degU, degI, E);
  k_weight<<<gE, TPB, 0, stream>>>(eSrc, eDst, degU, degI, wE, E);

  // layer-0 inputs + readout init
  k_init<<<gND, TPB, 0, stream>>>(usr_emb, itm_emb, XuA, XiA, outU, outI, ND);

  float *curU = XuA, *curI = XiA, *nxtU = XuB, *nxtI = XiB;
  for (int layer = 0; layer < 2; ++layer) {
    const float* Wa = W[2 * layer];
    const float* Wb = W[2 * layer + 1];

    k_cvt2<<<gDD, TPB, 0, stream>>>(Wa, Wb, WaBf, WbBf, D * D);
    k_cvt2<<<gND, TPB, 0, stream>>>(curU, curI, XuBf, XiBf, ND);

    // self/intra terms: Tu = Xu @ Wa^T, Ti = Xi @ Wa^T
    k_dense_wmma<<<gDense, TPB, 0, stream>>>(XuBf, WaBf, Tu, N);
    k_dense_wmma<<<gDense, TPB, 0, stream>>>(XiBf, WaBf, Ti, N);

    hipMemsetAsync(AggU, 0, (size_t)ND * 4, stream);
    hipMemsetAsync(AggI, 0, (size_t)ND * 4, stream);

    // user update: messages items(src=eDst) -> users(dst=eSrc), Tsrc = Ti
    k_edge_wmma<<<gEdge, TPB, 0, stream>>>(XiBf, XuBf, Ti, eDst, eSrc, wE,
                                           WbBf, AggU, E);
    // item update: messages users(src=eSrc) -> items(dst=eDst), Tsrc = Tu
    k_edge_wmma<<<gEdge, TPB, 0, stream>>>(XuBf, XiBf, Tu, eSrc, eDst, wE,
                                           WbBf, AggI, E);

    k_combine<<<gND, TPB, 0, stream>>>(Tu, AggU, nxtU, outU, ND);
    k_combine<<<gND, TPB, 0, stream>>>(Ti, AggI, nxtI, outI, ND);

    float* t;
    t = curU; curU = nxtU; nxtU = t;
    t = curI; curI = nxtI; nxtI = t;
  }
}
